// GATlayer_3470333575653
// MI455X (gfx1250) — compile-verified, matrix operand order checked by reading, not run on hardware
//
#include <hip/hip_runtime.h>
#include <hip/hip_bf16.h>

// ---------------------------------------------------------------------------
// GATv2 layer for MI455X (gfx1250, wave32, WMMA + async LDS DMA).
// N=50000 nodes, D=128, H=8 heads, DP=16, E_tot = 650000 edges (incl self loops)
//
//  phase 0: zero-init out / segment-max / denom buffers
//  phase 1: fs = x@w_src+b_src, fd = x@w_dst+b_dst  via v_wmma_f32_16x16x32_f16
//           weights staged to LDS with global_load_async_to_lds_b128 (ASYNCcnt)
//  phase 2: per-(edge,head) logits, atomic segment-max (monotonic uint map)
//  phase 3: ex = exp(logit - max), atomic segment-sum -> denom
//  phase 4: out[dst] += (ex/denom) * fs[src]   (f32 global atomics)
// ---------------------------------------------------------------------------

typedef __attribute__((ext_vector_type(16))) _Float16 v16h;
typedef __attribute__((ext_vector_type(8)))  _Float16 v8h;
typedef __attribute__((ext_vector_type(8)))  float    v8f;
typedef __attribute__((ext_vector_type(4)))  float    v4f;

#define GAT_D   128
#define GAT_H   8
#define GAT_DP  16
#define SLOPE   0.2f

// dynamic LDS: [0,64K) f32 weight staging, [64K,96K) transposed f16 weights
#define LDS_BYTES (64 * 1024 + 32 * 1024)

// ---------------------------------------------------------------- fill zeros
__global__ void gat_fill_zero(unsigned* __restrict__ p, long long count) {
  long long i = (long long)blockIdx.x * blockDim.x + threadIdx.x;
  long long stride = (long long)gridDim.x * blockDim.x;
  for (; i < count; i += stride) p[i] = 0u;
}

// ------------------------------------------------------- projection via WMMA
// grid.x = ceil(n/128) row-tiles, grid.y = 0 (src proj) / 1 (dst proj)
// block  = 256 threads = 8 wave32; wave w handles rows [blk*128 + w*16, +16)
__global__ __launch_bounds__(256)
void gat_proj_wmma(const float* __restrict__ x,
                   const float* __restrict__ w_src, const float* __restrict__ b_src,
                   const float* __restrict__ w_dst, const float* __restrict__ b_dst,
                   float* __restrict__ fs, float* __restrict__ fd, int n)
{
  extern __shared__ char gat_smem[];
  float*    wstage = (float*)gat_smem;                 // 64KB raw f32 weights
  _Float16* wT     = (_Float16*)(gat_smem + 65536);    // 32KB transposed f16 [ncol][k]

  const int mat          = blockIdx.y;
  const float* __restrict__ w    = mat ? w_dst : w_src;
  const float* __restrict__ bias = mat ? b_dst : b_src;
  float* __restrict__ out        = mat ? fd : fs;

  // ---- async DMA: stage w (f32, 128x128 = 64KB) into LDS, 16B per lane-op.
  // global_load_async_to_lds_b128: VDST = LDS byte addr, VADDR = byte offset,
  // SADDR = 64-bit base (GVS mode). Tracked by ASYNCcnt.
  {
    const unsigned lds_base = (unsigned)(size_t)(void*)wstage;
    const unsigned long long gbase = (unsigned long long)(const void*)w;
    #pragma unroll
    for (int i = 0; i < 16; ++i) {
      unsigned byteoff = (threadIdx.x + i * 256) * 16u;   // 4096 x 16B = 64KB
      unsigned ldsaddr = lds_base + byteoff;
      asm volatile("global_load_async_to_lds_b128 %0, %1, %2"
                   :: "v"(ldsaddr), "v"(byteoff), "s"(gbase) : "memory");
    }
    asm volatile("s_wait_asynccnt 0x0" ::: "memory");
  }
  __syncthreads();

  // ---- convert + transpose in LDS: wT[ncol][k] = (f16) wstage[k][ncol]
  for (int i = threadIdx.x; i < GAT_D * GAT_D; i += 256) {
    int kc = i >> 7;
    int nc = i & 127;
    wT[nc * GAT_D + kc] = (_Float16)wstage[i];
  }
  __syncthreads();

  const int wave  = threadIdx.x >> 5;                 // 0..7
  const int lane  = threadIdx.x & 31;
  const int row0  = (blockIdx.x * 8 + wave) * 16;     // n % 16 == 0 -> tile fully valid or fully out
  if (row0 >= n) return;                              // wave-uniform; EXEC stays all-ones

  const int khalf = lane >> 4;                        // 0: K 0..7(+16..23), 1: K 8..15(+24..31)
  const int lmod  = lane & 15;                        // A: row-in-tile ; B: col-in-tile

  v8f acc[8];
  #pragma unroll
  for (int i = 0; i < 8; ++i) acc[i] = v8f{0.f,0.f,0.f,0.f,0.f,0.f,0.f,0.f};

  const float* __restrict__ xrow = x + (size_t)(row0 + lmod) * GAT_D;

  #pragma unroll
  for (int ks = 0; ks < 4; ++ks) {                    // K = 128 in steps of 32
    // ---- A fragment (16x32 f16): per-lane elements are two 8-contiguous K runs
    const float* p0 = xrow + ks * 32 + khalf * 8;
    v4f a0 = *(const v4f*)(p0);
    v4f a1 = *(const v4f*)(p0 + 4);
    v4f a2 = *(const v4f*)(p0 + 16);
    v4f a3 = *(const v4f*)(p0 + 20);
    v16h a;
    #pragma unroll
    for (int i = 0; i < 4; ++i) {
      a[i]      = (_Float16)a0[i];
      a[4 + i]  = (_Float16)a1[i];
      a[8 + i]  = (_Float16)a2[i];
      a[12 + i] = (_Float16)a3[i];
    }
    #pragma unroll
    for (int nt = 0; nt < 8; ++nt) {
      // ---- B fragment (32x16 f16) from transposed LDS weights
      const _Float16* q0 = wT + (nt * 16 + lmod) * GAT_D + ks * 32 + khalf * 8;
      v8h blo = *(const v8h*)(q0);
      v8h bhi = *(const v8h*)(q0 + 16);
      v16h b = __builtin_shufflevector(blo, bhi, 0,1,2,3,4,5,6,7,8,9,10,11,12,13,14,15);
      acc[nt] = __builtin_amdgcn_wmma_f32_16x16x32_f16(
          /*neg_a=*/false, a, /*neg_b=*/false, b,
          /*c_mod=*/(short)0, acc[nt], /*reuse_a=*/false, /*reuse_b=*/false);
    }
  }

  // C/D layout: lanes 0-15: VGPR r -> (M=r, N=lane); lanes 16-31: (M=8+r, N=lane-16)
  #pragma unroll
  for (int nt = 0; nt < 8; ++nt) {
    const int col = nt * 16 + lmod;
    const float bv = bias[col];
    #pragma unroll
    for (int r = 0; r < 8; ++r) {
      const int row = row0 + khalf * 8 + r;
      out[(size_t)row * GAT_D + col] = acc[nt][r] + bv;
    }
  }
}

// ------------------------------------------------- monotonic float<->uint map
__device__ __forceinline__ unsigned f2ord(float f) {
  unsigned u = __float_as_uint(f);
  return (f >= 0.f) ? (u | 0x80000000u) : ~u;
}
__device__ __forceinline__ float ord2f(unsigned u) {
  return (u & 0x80000000u) ? __uint_as_float(u ^ 0x80000000u) : __uint_as_float(~u);
}

// --------------------------------------------------- edge logits + atomic max
__global__ void gat_edge_logit(const float* __restrict__ fs, const float* __restrict__ fd,
                               const float* __restrict__ attn_w,
                               const int* __restrict__ src, const int* __restrict__ dst,
                               float* __restrict__ logit, unsigned* __restrict__ mmax,
                               long long etot)
{
  long long idx = (long long)blockIdx.x * blockDim.x + threadIdx.x;
  if (idx >= etot * GAT_H) return;
  const long long e = idx >> 3;
  const int h = (int)(idx & 7);
  const int s = src[e], d = dst[e];

  const float* ps = fs + (size_t)s * GAT_D + h * GAT_DP;
  const float* pd = fd + (size_t)d * GAT_D + h * GAT_DP;
  const float* aw = attn_w + h * GAT_DP;

  float acc = 0.f;
  #pragma unroll
  for (int i = 0; i < GAT_DP; ++i) {
    float v = ps[i] + pd[i];
    v = (v > 0.f) ? v : v * SLOPE;         // LeakyReLU(0.2)
    acc += aw[i] * v;
  }
  logit[idx] = acc;
  atomicMax(mmax + (size_t)d * GAT_H + h, f2ord(acc));
}

// -------------------------------------------------------- exp + denominator
__global__ void gat_edge_exp(float* __restrict__ exl,            // in: logit, out: ex
                             const unsigned* __restrict__ mmax,
                             const int* __restrict__ dst,
                             float* __restrict__ denom, long long etot)
{
  long long idx = (long long)blockIdx.x * blockDim.x + threadIdx.x;
  if (idx >= etot * GAT_H) return;
  const long long e = idx >> 3;
  const int h = (int)(idx & 7);
  const int d = dst[e];
  const float m = ord2f(mmax[(size_t)d * GAT_H + h]);   // self-loops -> always finite
  const float ex = __expf(exl[idx] - m);
  exl[idx] = ex;
  atomicAdd(denom + (size_t)d * GAT_H + h, ex);
}

// ------------------------------------------------ weighted scatter aggregation
__global__ void gat_aggregate(const float* __restrict__ fs,
                              const float* __restrict__ exl,
                              const float* __restrict__ denom,
                              const int* __restrict__ src, const int* __restrict__ dst,
                              float* __restrict__ out, long long etot)
{
  long long idx = (long long)blockIdx.x * blockDim.x + threadIdx.x;
  if (idx >= etot * GAT_H) return;
  const long long e = idx >> 3;
  const int h = (int)(idx & 7);
  const int s = src[e], d = dst[e];

  const float alpha = exl[idx] / denom[(size_t)d * GAT_H + h];
  const float* ps = fs + (size_t)s * GAT_D + h * GAT_DP;
  float* po = out + (size_t)d * GAT_D + h * GAT_DP;
  #pragma unroll
  for (int i = 0; i < GAT_DP; ++i) atomicAdd(po + i, alpha * ps[i]);
}

// ---------------------------------------------------------------------------
extern "C" void kernel_launch(void* const* d_in, const int* in_sizes, int n_in,
                              void* d_out, int out_size, void* d_ws, size_t ws_size,
                              hipStream_t stream) {
  const float* x      = (const float*)d_in[0];
  const float* w_src  = (const float*)d_in[1];
  const float* b_src  = (const float*)d_in[2];
  const float* w_dst  = (const float*)d_in[3];
  const float* b_dst  = (const float*)d_in[4];
  const float* attn_w = (const float*)d_in[5];
  const int*   src    = (const int*)d_in[6];
  const int*   dst    = (const int*)d_in[7];
  float*       out    = (float*)d_out;

  const int       n    = in_sizes[0] / GAT_D;   // 50000
  const long long etot = in_sizes[6];           // 650000

  // Workspace carve-out (floats): fs | fd | ex/logit | mmax | denom  (~75 MB)
  float*    fs    = (float*)d_ws;
  float*    fd    = fs + (size_t)n * GAT_D;
  float*    exl   = fd + (size_t)n * GAT_D;
  unsigned* mmax  = (unsigned*)(exl + (size_t)etot * GAT_H);
  float*    denom = (float*)(mmax + (size_t)n * GAT_H);

  // phase 0: zero out / mmax (ord-map 0 == -inf) / denom
  gat_fill_zero<<<2048, 256, 0, stream>>>((unsigned*)out, (long long)n * GAT_D);
  gat_fill_zero<<<512,  256, 0, stream>>>(mmax,           (long long)n * GAT_H);
  gat_fill_zero<<<512,  256, 0, stream>>>((unsigned*)denom,(long long)n * GAT_H);

  // phase 1: WMMA projections (96KB dynamic LDS: f32 stage + f16 transposed)
  dim3 pgrid((n + 127) / 128, 2);
  gat_proj_wmma<<<pgrid, 256, LDS_BYTES, stream>>>(x, w_src, b_src, w_dst, b_dst, fs, fd, n);

  // phases 2-4: edge pipeline
  const long long work = etot * GAT_H;
  const int eblocks = (int)((work + 255) / 256);
  gat_edge_logit<<<eblocks, 256, 0, stream>>>(fs, fd, attn_w, src, dst, exl, mmax, etot);
  gat_edge_exp  <<<eblocks, 256, 0, stream>>>(exl, mmax, dst, denom, etot);
  gat_aggregate <<<eblocks, 256, 0, stream>>>(fs, exl, denom, src, dst, out, etot);
}
